// LayoutLMv2SelfAttention_15650860827273
// MI455X (gfx1250) — matrix operand order repaired and checked by reading, not compiled, not run on hardware
//
#include <hip/hip_runtime.h>

#define EMBED 1024
#define HEADS 16
#define HDIM  64
#define BATCH 2
#define SEQ   2048
#define BC    64   // keys per block in flash attention
#define AWAVES 8   // waves (16-row q tiles) per attention workgroup

typedef __attribute__((ext_vector_type(16))) _Float16 v16h;
typedef __attribute__((ext_vector_type(8)))  _Float16 v8h;
typedef __attribute__((ext_vector_type(8)))  float    v8f;
typedef __attribute__((ext_vector_type(4)))  float    v4f;

static __device__ __forceinline__ v8f wmma_f16(v16h a, v16h b, v8f c) {
  return __builtin_amdgcn_wmma_f32_16x16x32_f16(false, a, false, b, (short)0, c, false, false);
}

// Async 32-byte global->LDS copy (per lane), tracked by ASYNCcnt. [CDNA5]
static __device__ __forceinline__ void async_copy32(unsigned int lds, unsigned long long g) {
  asm volatile(
      "global_load_async_to_lds_b128 %0, %1, off\n\t"
      "global_load_async_to_lds_b128 %0, %1, off offset:16"
      :: "v"(lds), "v"(g) : "memory");
}
static __device__ __forceinline__ void wait_async0() {
  asm volatile("s_wait_asynccnt 0x0" ::: "memory");
}

// A fragment (16x32 f16): lane holds row r=lane&15; half=lane>>4.
// a[j] = A[r][k0 + half*16 + j], a[8+j] = A[r][k0 + half*16 + 8 + j]
static __device__ __forceinline__ v16h load_afrag_f16(const _Float16* rowp, int k0, int half) {
  v8h x0 = *(const v8h*)(rowp + k0 + half * 16);
  v8h x1 = *(const v8h*)(rowp + k0 + half * 16 + 8);
  v16h a;
#pragma unroll
  for (int j = 0; j < 8; ++j) { a[j] = x0[j]; a[8 + j] = x1[j]; }
  return a;
}

// B fragment: lane holds column n=lane&15, half=lane>>4.
// b[j] = B[K = half*16 + j][n]; caller passes p = &elem(K=half*16, n), contiguous in K.
static __device__ __forceinline__ v16h load_bfrag_f16(const _Float16* p) {
  v8h x0 = *(const v8h*)p;
  v8h x1 = *(const v8h*)(p + 8);
  v16h b;
#pragma unroll
  for (int j = 0; j < 8; ++j) { b[j] = x0[j]; b[8 + j] = x1[j]; }
  return b;
}

// One-shot fp32 -> f16 conversion (X and the four weight matrices). n % 8 == 0.
__global__ __launch_bounds__(256) void cvt_kernel(const float* __restrict__ in,
                                                  _Float16* __restrict__ out, int n) {
  const int idx = (blockIdx.x * 256 + threadIdx.x) * 8;
  if (idx >= n) return;
  const v4f* p = (const v4f*)(in + idx);
  v4f a = p[0], b = p[1];
  v8h o;
#pragma unroll
  for (int j = 0; j < 4; ++j) { o[j] = (_Float16)a[j]; o[4 + j] = (_Float16)b[j]; }
  *(v8h*)(out + idx) = o;
}

// C = A[4096x1024](f16) * W^T[1024x1024](f16) (+bias)*scale. One 16x64 tile/wave.
// HEADSPLIT: write f16 [B,H,S,Dh]; else f32 [M,N] to d_out.
template <bool HEADSPLIT>
__global__ __launch_bounds__(128) void proj_kernel(const _Float16* __restrict__ A,
                                                   const _Float16* __restrict__ W,
                                                   const float* __restrict__ bias,
                                                   void* __restrict__ out_,
                                                   float scale) {
  const int lane  = threadIdx.x & 31;
  const int wave  = threadIdx.x >> 5;
  const int wid   = blockIdx.x * 4 + wave;   // 0..4095
  const int mtile = wid >> 4;                // 0..255
  const int ncol  = (wid & 15) * 64;         // 0..960
  const int r     = lane & 15;
  const int half  = lane >> 4;

  const v8f vz = {0.f,0.f,0.f,0.f,0.f,0.f,0.f,0.f};
  v8f acc0 = vz, acc1 = vz, acc2 = vz, acc3 = vz;

  const _Float16* arow = A + (size_t)(mtile * 16 + r) * EMBED;
  const _Float16* w0 = W + (size_t)(ncol +  0 + r) * EMBED;
  const _Float16* w1 = W + (size_t)(ncol + 16 + r) * EMBED;
  const _Float16* w2 = W + (size_t)(ncol + 32 + r) * EMBED;
  const _Float16* w3 = W + (size_t)(ncol + 48 + r) * EMBED;

  for (int k0 = 0; k0 < EMBED; k0 += 32) {
    v16h a = load_afrag_f16(arow, k0, half);
    acc0 = wmma_f16(a, load_bfrag_f16(w0 + k0 + half * 16), acc0);
    acc1 = wmma_f16(a, load_bfrag_f16(w1 + k0 + half * 16), acc1);
    acc2 = wmma_f16(a, load_bfrag_f16(w2 + k0 + half * 16), acc2);
    acc3 = wmma_f16(a, load_bfrag_f16(w3 + k0 + half * 16), acc3);
  }

  // C layout: VGPR i -> row = mtile*16 + half*8 + i, col = ncol + (lane&15) + 16*nt
  const int col0 = ncol + r;
  const int rowb = mtile * 16 + half * 8;
#pragma unroll
  for (int i = 0; i < 8; ++i) {
    const int row = rowb + i;
    float v0 = (acc0[i] + bias[col0 +  0]) * scale;
    float v1 = (acc1[i] + bias[col0 + 16]) * scale;
    float v2 = (acc2[i] + bias[col0 + 32]) * scale;
    float v3 = (acc3[i] + bias[col0 + 48]) * scale;
    if (HEADSPLIT) {
      _Float16* out = (_Float16*)out_;
      const int bb = row >> 11, s = row & (SEQ - 1);
      auto put = [&](int col, float v) {
        int h = col >> 6, dh = col & (HDIM - 1);
        out[(((size_t)(bb * HEADS + h) * SEQ) + s) * HDIM + dh] = (_Float16)v;
      };
      put(col0, v0); put(col0 + 16, v1); put(col0 + 32, v2); put(col0 + 48, v3);
    } else {
      float* out = (float*)out_;
      out[(size_t)row * EMBED + col0 +  0] = v0;
      out[(size_t)row * EMBED + col0 + 16] = v1;
      out[(size_t)row * EMBED + col0 + 32] = v2;
      out[(size_t)row * EMBED + col0 + 48] = v3;
    }
  }
}

// Flash attention: grid (B*H, S/128), 256 threads = 8 waves; wave = one 16-query tile.
// Q pre-scaled by 1/sqrt(Dh). 64-key blocks, double-buffered LDS tiles shared by 8 waves:
// K streams in via GLOBAL_LOAD_ASYNC_TO_LDS (ASYNCcnt), V via VGPR transpose.
// Row-sums via WMMA (P x ones); row-max butterfly batched: 8 bpermutes per step, one wait.
__global__ __launch_bounds__(256) void attn_kernel(const _Float16* __restrict__ Q,
                                                   const _Float16* __restrict__ K,
                                                   const _Float16* __restrict__ V,
                                                   const float* __restrict__ bias,
                                                   _Float16* __restrict__ ctx) {
  __shared__ _Float16 Kt[2][BC * HDIM];        // [key][d]   2 x 8 KB
  __shared__ _Float16 Vt[2][HDIM * BC];        // [d][key]   2 x 8 KB (transposed)
  __shared__ _Float16 Pb[AWAVES][16 * BC];     // per-wave P 16 KB

  const int tid  = threadIdx.x;
  const int lane = tid & 31, wave = tid >> 5;
  const int bh = blockIdx.x;
  const int b  = bh >> 4, h = bh & (HEADS - 1);
  const int q0 = blockIdx.y * (16 * AWAVES) + wave * 16;
  const int r = lane & 15, half = lane >> 4;

  const _Float16* Qb = Q + ((size_t)bh * SEQ + q0) * HDIM;
  const _Float16* Kb = K + (size_t)bh * SEQ * HDIM;
  const _Float16* Vb = V + (size_t)bh * SEQ * HDIM;
  const float* brow  = bias + (size_t)h * SEQ * SEQ;

  // K async staging: 256 threads x 32B = 8KB tile
  const unsigned long long kgbase = (unsigned long long)(uintptr_t)Kb + (unsigned long long)tid * 32ull;
  const unsigned int ldsK0 = (unsigned int)(uintptr_t)(&Kt[0][0]) + tid * 32;
  const unsigned int ldsK1 = (unsigned int)(uintptr_t)(&Kt[1][0]) + tid * 32;

  // V staging: key = tid&63, d range vdg*16..+15 (16 halfs per thread)
  const int vkey = tid & 63, vdg = tid >> 6;

  // Q A-fragments for the two d-steps (Dh=64 = 2 x K32)
  v16h qa0 = load_afrag_f16(Qb + r * HDIM, 0, half);
  v16h qa1 = load_afrag_f16(Qb + r * HDIM, 32, half);

  // all-ones B fragment for WMMA row-sum
  v16h bones;
#pragma unroll
  for (int j = 0; j < 16; ++j) bones[j] = (_Float16)1.0f;

  const v8f vz = {0.f,0.f,0.f,0.f,0.f,0.f,0.f,0.f};
  v8f o0 = vz, o1 = vz, o2 = vz, o3 = vz;
  float m[8], l[8];
#pragma unroll
  for (int i = 0; i < 8; ++i) { m[i] = -1e30f; l[i] = 0.f; }

  // ---- prologue: stage tile 0 into buffer 0 ----
  async_copy32(ldsK0, kgbase);
  {
    const v8h* vp = (const v8h*)(Vb + (size_t)vkey * HDIM + vdg * 16);
    v8h y0 = vp[0], y1 = vp[1];
#pragma unroll
    for (int j = 0; j < 8; ++j) {
      Vt[0][(vdg * 16 + j) * BC + vkey]     = y0[j];
      Vt[0][(vdg * 16 + 8 + j) * BC + vkey] = y1[j];
    }
  }
  wait_async0();
  __syncthreads();

  for (int kk = 0; kk < SEQ; kk += BC) {
    const int cur = (kk >> 6) & 1;
    const int nxt = cur ^ 1;
    const bool more = (kk + BC) < SEQ;

    // kick off next K tile into the other buffer (safe: last read in the previous iteration, before its barrier)
    if (more) async_copy32(cur ? ldsK0 : ldsK1, kgbase + (unsigned long long)(kk + BC) * (HDIM * 2));

    // issue next V tile's global loads now; LDS stores deferred to iteration end
    v8h vy0, vy1;
    if (more) {
      const v8h* vp = (const v8h*)(Vb + (size_t)(kk + BC + vkey) * HDIM + vdg * 16);
      vy0 = vp[0]; vy1 = vp[1];
    }

    // Scores: S[16q x 64keys] = Q (16x64) * K^T (64x64), 4 n-tiles x 2 d-steps
    v8f s0 = vz, s1 = vz, s2 = vz, s3 = vz;
    {
      const _Float16* kt0 = Kt[cur] + (0  + r) * HDIM;
      const _Float16* kt1 = Kt[cur] + (16 + r) * HDIM;
      const _Float16* kt2 = Kt[cur] + (32 + r) * HDIM;
      const _Float16* kt3 = Kt[cur] + (48 + r) * HDIM;
      s0 = wmma_f16(qa0, load_bfrag_f16(kt0 + half * 16), s0);
      s0 = wmma_f16(qa1, load_bfrag_f16(kt0 + 32 + half * 16), s0);
      s1 = wmma_f16(qa0, load_bfrag_f16(kt1 + half * 16), s1);
      s1 = wmma_f16(qa1, load_bfrag_f16(kt1 + 32 + half * 16), s1);
      s2 = wmma_f16(qa0, load_bfrag_f16(kt2 + half * 16), s2);
      s2 = wmma_f16(qa1, load_bfrag_f16(kt2 + 32 + half * 16), s2);
      s3 = wmma_f16(qa0, load_bfrag_f16(kt3 + half * 16), s3);
      s3 = wmma_f16(qa1, load_bfrag_f16(kt3 + 32 + half * 16), s3);
    }

    // Bias add (into s* in place) + per-row local max.
    float mx[8];
#pragma unroll
    for (int i = 0; i < 8; ++i) {
      const float* bp = brow + (size_t)(q0 + half * 8 + i) * SEQ + kk + r;
      if (more) __builtin_prefetch(bp + BC, 0, 3);  // next block's bias line, near-cache
      float t0 = s0[i] + bp[0];
      float t1 = s1[i] + bp[16];
      float t2 = s2[i] + bp[32];
      float t3 = s3[i] + bp[48];
      s0[i] = t0; s1[i] = t1; s2[i] = t2; s3[i] = t3;
      mx[i] = fmaxf(fmaxf(t0, t1), fmaxf(t2, t3));
    }
    // Batched butterfly max over the 16-lane row group: 8 bpermutes/step, one wait.
#pragma unroll
    for (int off = 8; off >= 1; off >>= 1) {
      float tmp[8];
#pragma unroll
      for (int i = 0; i < 8; ++i) tmp[i] = __shfl_xor(mx[i], off, 32);
#pragma unroll
      for (int i = 0; i < 8; ++i) mx[i] = fmaxf(mx[i], tmp[i]);
    }

    // Online softmax finalize per row; write P (f16) to wave-private LDS scratch.
    _Float16* pw = Pb[wave];
#pragma unroll
    for (int i = 0; i < 8; ++i) {
      float mn = fmaxf(m[i], mx[i]);
      float corr = __expf(m[i] - mn);
      m[i] = mn;
      l[i] *= corr;
      o0[i] *= corr; o1[i] *= corr; o2[i] *= corr; o3[i] *= corr;
      const int prow = half * 8 + i;
      pw[prow * BC +  0 + r] = (_Float16)__expf(s0[i] - mn);
      pw[prow * BC + 16 + r] = (_Float16)__expf(s1[i] - mn);
      pw[prow * BC + 32 + r] = (_Float16)__expf(s2[i] - mn);
      pw[prow * BC + 48 + r] = (_Float16)__expf(s3[i] - mn);
    }

    // P in A layout for both the row-sum WMMA and P*V (wave-private LDS, DS in-order)
    v16h pa0 = load_afrag_f16(pw + r * BC, 0, half);
    v16h pa1 = load_afrag_f16(pw + r * BC, 32, half);

    // l_blk = P x ones: every C column holds the row sum; element i = row half*8+i.
    v8f lacc = wmma_f16(pa0, bones, vz);
    lacc = wmma_f16(pa1, bones, lacc);
#pragma unroll
    for (int i = 0; i < 8; ++i) l[i] += lacc[i];

    // ctx += P (16x64) * V (64x64)
    const _Float16* vt = Vt[cur];
    o0 = wmma_f16(pa0, load_bfrag_f16(vt + (0  + r) * BC + half * 16), o0);
    o0 = wmma_f16(pa1, load_bfrag_f16(vt + (0  + r) * BC + 32 + half * 16), o0);
    o1 = wmma_f16(pa0, load_bfrag_f16(vt + (16 + r) * BC + half * 16), o1);
    o1 = wmma_f16(pa1, load_bfrag_f16(vt + (16 + r) * BC + 32 + half * 16), o1);
    o2 = wmma_f16(pa0, load_bfrag_f16(vt + (32 + r) * BC + half * 16), o2);
    o2 = wmma_f16(pa1, load_bfrag_f16(vt + (32 + r) * BC + 32 + half * 16), o2);
    o3 = wmma_f16(pa0, load_bfrag_f16(vt + (48 + r) * BC + half * 16), o3);
    o3 = wmma_f16(pa1, load_bfrag_f16(vt + (48 + r) * BC + 32 + half * 16), o3);

    // commit next V tile into the other buffer, wait K async, then rendezvous
    if (more) {
#pragma unroll
      for (int j = 0; j < 8; ++j) {
        Vt[nxt][(vdg * 16 + j) * BC + vkey]     = vy0[j];
        Vt[nxt][(vdg * 16 + 8 + j) * BC + vkey] = vy1[j];
      }
    }
    wait_async0();
    __syncthreads();
  }

  // Normalize and store ctx as f16 [B, S, H*Dh] (matches transpose+reshape)
#pragma unroll
  for (int i = 0; i < 8; ++i) {
    const float inv = 1.f / l[i];
    const int srow = q0 + half * 8 + i;
    const size_t base = ((size_t)b * SEQ + srow) * EMBED + h * HDIM + r;
    ctx[base +  0] = (_Float16)(o0[i] * inv);
    ctx[base + 16] = (_Float16)(o1[i] * inv);
    ctx[base + 32] = (_Float16)(o2[i] * inv);
    ctx[base + 48] = (_Float16)(o3[i] * inv);
  }
}

extern "C" void kernel_launch(void* const* d_in, const int* in_sizes, int n_in,
                              void* d_out, int out_size, void* d_ws, size_t ws_size,
                              hipStream_t stream) {
  (void)in_sizes; (void)n_in; (void)out_size; (void)ws_size;
  const float* X  = (const float*)d_in[0];
  const float* Ab = (const float*)d_in[1];
  const float* Wq = (const float*)d_in[2];
  const float* bq = (const float*)d_in[3];
  const float* Wk = (const float*)d_in[4];
  const float* bk = (const float*)d_in[5];
  const float* Wv = (const float*)d_in[6];
  const float* bv = (const float*)d_in[7];
  const float* Wo = (const float*)d_in[8];
  const float* bo = (const float*)d_in[9];

  const size_t HS = (size_t)BATCH * HEADS * SEQ * HDIM;  // 4,194,304 elems
  const size_t WN = (size_t)EMBED * EMBED;               // 1,048,576 elems
  _Float16* qh  = (_Float16*)d_ws;
  _Float16* kh  = qh + HS;
  _Float16* vh  = kh + HS;
  _Float16* x16 = vh + HS;          // f16 copy of X; later reused as ctx (aliased)
  _Float16* cx  = x16;              // ctx written after X's last use (proj V)
  _Float16* wq16 = x16 + HS;
  _Float16* wk16 = wq16 + WN;
  _Float16* wv16 = wk16 + WN;
  _Float16* wo16 = wv16 + WN;       // total ws use = 40 MiB

  // one-shot fp32 -> f16 conversions (X + 4 weight matrices)
  cvt_kernel<<<(int)(HS / 2048), 256, 0, stream>>>(X,  x16,  (int)HS);
  cvt_kernel<<<(int)(WN / 2048), 256, 0, stream>>>(Wq, wq16, (int)WN);
  cvt_kernel<<<(int)(WN / 2048), 256, 0, stream>>>(Wk, wk16, (int)WN);
  cvt_kernel<<<(int)(WN / 2048), 256, 0, stream>>>(Wv, wv16, (int)WN);
  cvt_kernel<<<(int)(WN / 2048), 256, 0, stream>>>(Wo, wo16, (int)WN);

  dim3 blk(128);
  proj_kernel<true><<<1024, blk, 0, stream>>>(x16, wq16, bq, qh, 0.125f);  // 1/sqrt(64)
  proj_kernel<true><<<1024, blk, 0, stream>>>(x16, wk16, bk, kh, 1.0f);
  proj_kernel<true><<<1024, blk, 0, stream>>>(x16, wv16, bv, vh, 1.0f);
  attn_kernel<<<dim3(BATCH * HEADS, SEQ / (16 * AWAVES)), 256, 0, stream>>>(qh, kh, vh, Ab, cx);
  proj_kernel<false><<<1024, blk, 0, stream>>>(cx, wo16, bo, d_out, 1.0f);
}
